// omninet_early_stop_14663018348603
// MI455X (gfx1250) — matrix operand order repaired and checked by reading, not compiled, not run on hardware
//
#include <hip/hip_runtime.h>
#include <hip/hip_bf16.h>

// ---------------- constants ----------------
#define C_T    44
#define C_NPIX 49
#define C_DM   500
#define C_H    8
#define C_DFC  100
#define C_NC   24
#define C_NA   2

typedef __attribute__((ext_vector_type(16))) _Float16 v16h;
typedef __attribute__((ext_vector_type(8)))  float    v8f;

// ---------------- WMMA fragment helpers (wave32, 16x16x32 f16) ----------------
// A-matrix 16x32 f16 layout, LDS row-major [m][k]:
//   lane<16 -> row=lane,    K pairs {0..7,16..23}
//   lane>=16 -> row=lane-16, K pairs {8..15,24..31}
// Two aligned 16-byte runs -> 2x ds_load_b128.
__device__ __forceinline__ v16h load_a_frag(const _Float16* lds, int ldl,
                                            int m0, int k0, int lane) {
  const int m  = m0 + (lane & 15);
  const int kb = k0 + ((lane & 16) ? 8 : 0);
  const _Float16* p = lds + (size_t)m * ldl + kb;
  union { v16h v; uint4 q[2]; } u;
  u.q[0] = *(const uint4*)(p);        // halves 0..7   (K = kb..kb+7)
  u.q[1] = *(const uint4*)(p + 16);   // halves 8..15  (K = kb+16..kb+23)
  return u.v;
}

// B-matrix 32x16 f16, LDS stored column-major: lds[n][k] == B[k][n].
//   lane<16 -> col=lane,    K {0..7,16..23} ; lane>=16 -> col=lane-16, K {8..15,24..31}
// Two aligned 16-byte runs -> 2x ds_load_b128.
__device__ __forceinline__ v16h load_b_frag_t(const _Float16* lds, int ldl,
                                              int k0, int n0, int lane) {
  const int n  = n0 + (lane & 15);
  const int kb = k0 + ((lane & 16) ? 8 : 0);
  const _Float16* p = lds + (size_t)n * ldl + kb;
  union { v16h v; uint4 q[2]; } u;
  u.q[0] = *(const uint4*)(p);
  u.q[1] = *(const uint4*)(p + 16);
  return u.v;
}

__device__ __forceinline__ uint2 pack4h(float4 v4) {
  union { uint2 u; _Float16 h[4]; } p;
  p.h[0] = (_Float16)v4.x; p.h[1] = (_Float16)v4.y;
  p.h[2] = (_Float16)v4.z; p.h[3] = (_Float16)v4.w;
  return p.u;
}

// ---------------- generic tiled WMMA GEMM ----------------
// C[M,N] = A[M,K] (f32) x W[N,K]^T (f32) + bias, optional ReLU.
// Block: 256 threads = 8 waves; tile 64x64; wave (wm 0..3, wn 0..1) -> 16x32.
#define BM 64
#define BN 64
#define BKK 32

__global__ __launch_bounds__(256) void gemm_wmma_kernel(
    const float* __restrict__ A, const float* __restrict__ W,
    const float* __restrict__ bias, float* __restrict__ C,
    int M, int N, int K, int relu)
{
  __shared__ _Float16 sA[BM][BKK];   // [m][k] row-major
  __shared__ _Float16 sB[BN][BKK];   // [n][k] fragment-ready (== B[k][n])

  const int tid  = threadIdx.x;
  const int lane = tid & 31;
  const int wave = tid >> 5;
  const int wm   = wave >> 1;   // 0..3
  const int wn   = wave & 1;    // 0..1
  const int m0   = blockIdx.y * BM;
  const int n0   = blockIdx.x * BN;

  // per-thread tile-load coordinates (k-invariant parts hoisted)
  const int ra  = (tid >> 3) & 63;            // A row within tile (i=0 slice)
  const int ca  = (tid & 7) << 2;             // A k4 within tile
  const int gmA0 = m0 + ra;
  const int gmA1 = m0 + ((tid + 256) >> 3 & 63);
  const int caA1 = ((tid + 256) & 7) << 2;

  v8f acc0 = {}; v8f acc1 = {};

  for (int k0 = 0; k0 < K; k0 += BKK) {
    // ---- A tile: 64x32 f32 = 512 float4, 2 per thread; branch-free guards ----
#pragma unroll
    for (int i = 0; i < 2; ++i) {
      const int f  = tid + i * 256;
      const int r  = f >> 3, c4 = (f & 7) << 2;
      const int gm = m0 + r, gk = k0 + c4;
      const int gmc = gm < M ? gm : M - 1;
      const int gkc = gk + 3 < K ? gk : K - 4;
      float4 v4 = *(const float4*)(A + (size_t)gmc * K + gkc);
      if (gm >= M || gk + 3 >= K) v4 = make_float4(0.f, 0.f, 0.f, 0.f);
      *(uint2*)&sA[r][c4] = pack4h(v4);
      if (k0 + BKK < K) __builtin_prefetch(A + (size_t)gmc * K + gkc + BKK, 0, 3);
    }
    // ---- W tile -> sB[n][k]; W rows are contiguous in K ----
#pragma unroll
    for (int i = 0; i < 2; ++i) {
      const int f  = tid + i * 256;
      const int n  = f >> 3, k4 = (f & 7) << 2;
      const int gn = n0 + n, gk = k0 + k4;
      const int gnc = gn < N ? gn : N - 1;
      const int gkc = gk + 3 < K ? gk : K - 4;
      float4 v4 = *(const float4*)(W + (size_t)gnc * K + gkc);
      if (gn >= N || gk + 3 >= K) v4 = make_float4(0.f, 0.f, 0.f, 0.f);
      *(uint2*)&sB[n][k4] = pack4h(v4);
      if (k0 + BKK < K) __builtin_prefetch(W + (size_t)gnc * K + gkc + BKK, 0, 3);
    }
    __syncthreads();

    v16h a  = load_a_frag(&sA[0][0], BKK, wm * 16, 0, lane);
    v16h b0 = load_b_frag_t(&sB[0][0], BKK, 0, wn * 32,      lane);
    v16h b1 = load_b_frag_t(&sB[0][0], BKK, 0, wn * 32 + 16, lane);
    acc0 = __builtin_amdgcn_wmma_f32_16x16x32_f16(false, a, false, b0, (short)0, acc0, false, false);
    acc1 = __builtin_amdgcn_wmma_f32_16x16x32_f16(false, a, false, b1, (short)0, acc1, false, false);
    __syncthreads();
  }

  const int rbase = m0 + wm * 16 + ((lane & 16) ? 8 : 0);
#pragma unroll
  for (int t = 0; t < 2; ++t) {
    v8f acc = t ? acc1 : acc0;
    const int col = n0 + wn * 32 + t * 16 + (lane & 15);
    if (col < N) {
      const float bv = bias ? bias[col] : 0.f;
#pragma unroll
      for (int j = 0; j < 8; ++j) {
        const int row = rbase + j;
        if (row < M) {
          float v = acc[j] + bv;
          if (relu) v = fmaxf(v, 0.f);
          C[(size_t)row * N + col] = v;
        }
      }
    }
  }
}

// ---------------- image mean over 49 pixels ----------------
__global__ __launch_bounds__(128) void img_mean_kernel(
    const float* __restrict__ spatial, float* __restrict__ mean)
{
  const int b = blockIdx.x;
  for (int d = threadIdx.x; d < C_DM; d += 128) {
    float s = 0.f;
    for (int p = 0; p < C_NPIX; ++p)
      s += spatial[((size_t)b * C_NPIX + p) * C_DM + d];
    mean[(size_t)b * C_DM + d] = s * (1.f / (float)C_NPIX);
  }
}

// ---------------- assemble time_cache (+ sinusoidal positions) ----------------
__global__ __launch_bounds__(128) void assemble_tc_kernel(
    const float* __restrict__ th, const float* __restrict__ ta,
    const float* __restrict__ tm, const float* __restrict__ tcap,
    const float* __restrict__ tb, float* __restrict__ tc)
{
  const int row = blockIdx.x;          // b*T + t
  const int b = row / C_T, t = row % C_T;
  const float* src;
  if      (t == 0) src = th   + (size_t)b * C_DM;
  else if (t == 1) src = ta   + (size_t)b * C_DM;
  else if (t == 2) src = tm   + (size_t)b * C_DM;
  else if (t == 3) src = tcap + (size_t)b * C_DM;
  else             src = tb   + ((size_t)b * 40 + (t - 4)) * C_DM;
  for (int d = threadIdx.x; d < C_DM; d += 128) {
    const float expo  = (float)(2 * (d >> 1)) / (float)C_DM;
    const float ang   = (float)t / powf(10000.f, expo);
    const float pos   = (d & 1) ? cosf(ang) : sinf(ang);
    tc[(size_t)row * C_DM + d] = src[d] + pos;
  }
}

// ---------------- residual add + layernorm ----------------
__global__ __launch_bounds__(128) void add_ln_kernel(
    const float* __restrict__ X, const float* __restrict__ R,
    const float* __restrict__ g, const float* __restrict__ be,
    float* __restrict__ O)
{
  const int row = blockIdx.x;
  const int tid = threadIdx.x;
  __shared__ float rs[128], rq[128];
  float xv[4];
  float s = 0.f, q = 0.f; int cnt = 0;
  for (int d = tid; d < C_DM; d += 128) {
    float v = X[(size_t)row * C_DM + d] + R[(size_t)row * C_DM + d];
    xv[cnt++] = v; s += v; q += v * v;
  }
  rs[tid] = s; rq[tid] = q; __syncthreads();
  for (int off = 64; off; off >>= 1) {
    if (tid < off) { rs[tid] += rs[tid + off]; rq[tid] += rq[tid + off]; }
    __syncthreads();
  }
  const float mu  = rs[0] * (1.f / (float)C_DM);
  const float var = rq[0] * (1.f / (float)C_DM) - mu * mu;
  const float inv = rsqrtf(var + 1e-5f);
  cnt = 0;
  for (int d = tid; d < C_DM; d += 128)
    O[(size_t)row * C_DM + d] = (xv[cnt++] - mu) * inv * g[d] + be[d];
}

// ---------------- helpers for attention tile loads ----------------
// Load rows [0..rows) of a strided f32 matrix (row stride 512, 64 cols) into
// LDS row-major f16 [ldRows][64]; rows >= valid are zero-filled. Branch-free.
__device__ __forceinline__ void load_rows_f16(
    const float* __restrict__ src, _Float16* __restrict__ dst,
    int ldRows, int valid, int lane)
{
  for (int e = lane; e < ldRows * 16; e += 32) {
    const int r = e >> 4, c4 = (e & 15) << 2;
    const int rc = r < valid ? r : valid - 1;
    float4 v4 = *(const float4*)(src + (size_t)rc * 512 + c4);
    if (r >= valid) v4 = make_float4(0.f, 0.f, 0.f, 0.f);
    *(uint2*)(dst + (size_t)r * 64 + c4) = pack4h(v4);
  }
}

// Load V (valid rows of 64 feats, stride 512) transposed into LDS [feat][key].
__device__ __forceinline__ void load_rows_f16_T(
    const float* __restrict__ src, _Float16* __restrict__ dst,
    int ldKeys, int valid, int lane)
{
  for (int e = lane; e < ldKeys * 16; e += 32) {
    const int r = e >> 4, c4 = (e & 15) << 2;   // r = key, c4 = feat group
    const int rc = r < valid ? r : valid - 1;
    float4 v4 = *(const float4*)(src + (size_t)rc * 512 + c4);
    if (r >= valid) v4 = make_float4(0.f, 0.f, 0.f, 0.f);
    dst[(size_t)(c4 + 0) * ldKeys + r] = (_Float16)v4.x;
    dst[(size_t)(c4 + 1) * ldKeys + r] = (_Float16)v4.y;
    dst[(size_t)(c4 + 2) * ldKeys + r] = (_Float16)v4.z;
    dst[(size_t)(c4 + 3) * ldKeys + r] = (_Float16)v4.w;
  }
}

// ---------------- self-attention (causal) per (b,h), 1 wave32 ----------------
__global__ __launch_bounds__(32) void attn_self_kernel(
    const float* __restrict__ Q, const float* __restrict__ K,
    const float* __restrict__ V, float* __restrict__ O,
    float* __restrict__ gate)
{
  const int lane = threadIdx.x;
  const int bh = blockIdx.x;
  const int b = bh >> 3, h = bh & 7;
  const size_t base = ((size_t)b * C_T) * 512 + (size_t)h * 64;

  __shared__ _Float16 sQ[48][64];    // A operand rows (queries)
  __shared__ _Float16 sK[48][64];    // B operand, fragment-ready: B[k][n]=sK[key][feat]
  __shared__ _Float16 sVT[64][64];   // B operand, fragment-ready: B[key][feat]=sVT[feat][key]
  __shared__ float    sS[48][48];
  __shared__ _Float16 sP[48][64];    // A operand rows (probabilities)

  load_rows_f16(Q + base, &sQ[0][0], 48, C_T, lane);
  load_rows_f16(K + base, &sK[0][0], 48, C_T, lane);
  load_rows_f16_T(V + base, &sVT[0][0], 64, C_T, lane);
  __syncthreads();

  // S = (Q x K^T) * scale  -- contraction over 64 feats
#pragma unroll
  for (int mt = 0; mt < 3; ++mt)
#pragma unroll
    for (int nt = 0; nt < 3; ++nt) {
      v8f acc = {};
#pragma unroll
      for (int ks = 0; ks < 2; ++ks) {
        v16h a  = load_a_frag(&sQ[0][0], 64, mt * 16, ks * 32, lane);
        v16h bb = load_b_frag_t(&sK[0][0], 64, ks * 32, nt * 16, lane);
        acc = __builtin_amdgcn_wmma_f32_16x16x32_f16(false, a, false, bb, (short)0, acc, false, false);
      }
      const int c = nt * 16 + (lane & 15);
      const int rb = mt * 16 + ((lane & 16) ? 8 : 0);
#pragma unroll
      for (int j = 0; j < 8; ++j) sS[rb + j][c] = acc[j] * 0.125f;
    }
  __syncthreads();

  // causal softmax per row; emit gate = P[:, IMG_ARR=2]
  for (int r = lane; r < 48; r += 32) {
    if (r < C_T) {
      float mx = -3.4e38f;
      for (int j = 0; j <= r; ++j) mx = fmaxf(mx, sS[r][j]);
      float sum = 0.f;
      for (int j = 0; j <= r; ++j) { float e = __expf(sS[r][j] - mx); sS[r][j] = e; sum += e; }
      const float inv = 1.f / sum;
      for (int j = 0; j < 64; ++j)
        sP[r][j] = (j <= r) ? (_Float16)(sS[r][j] * inv) : (_Float16)0.f;
      gate[(size_t)bh * C_T + r] = (r >= 2) ? sS[r][2] * inv : 0.f;
    } else {
      for (int j = 0; j < 64; ++j) sP[r][j] = (_Float16)0.f;
    }
  }
  __syncthreads();

  // O = P x V  -- contraction over 64 keys (>=44 are zero)
#pragma unroll
  for (int mt = 0; mt < 3; ++mt)
#pragma unroll
    for (int nt = 0; nt < 4; ++nt) {
      v8f acc = {};
#pragma unroll
      for (int ks = 0; ks < 2; ++ks) {
        v16h a  = load_a_frag(&sP[0][0], 64, mt * 16, ks * 32, lane);
        v16h bb = load_b_frag_t(&sVT[0][0], 64, ks * 32, nt * 16, lane);
        acc = __builtin_amdgcn_wmma_f32_16x16x32_f16(false, a, false, bb, (short)0, acc, false, false);
      }
      const int c = nt * 16 + (lane & 15);
      const int rb = mt * 16 + ((lane & 16) ? 8 : 0);
#pragma unroll
      for (int j = 0; j < 8; ++j) {
        const int r = rb + j;
        if (r < C_T) O[base + (size_t)r * 512 + c] = acc[j];
      }
    }
}

// ---------------- gated cross-attention per (b,h), 1 wave32 ----------------
__global__ __launch_bounds__(32) void attn_cross_kernel(
    const float* __restrict__ Q, const float* __restrict__ K,
    const float* __restrict__ V, const float* __restrict__ gate,
    float* __restrict__ O)
{
  const int lane = threadIdx.x;
  const int bh = blockIdx.x;
  const int b = bh >> 3, h = bh & 7;
  const size_t qbase  = ((size_t)b * C_T)    * 512 + (size_t)h * 64;
  const size_t kvbase = ((size_t)b * C_NPIX) * 512 + (size_t)h * 64;

  __shared__ _Float16 sQ[48][64];
  __shared__ _Float16 sK[64][64];    // keys 49..63 zero
  __shared__ _Float16 sVT[64][64];   // [feat][key], keys 49..63 zero
  __shared__ float    sS[48][64];
  __shared__ _Float16 sP[48][64];

  load_rows_f16(Q + qbase, &sQ[0][0], 48, C_T, lane);
  load_rows_f16(K + kvbase, &sK[0][0], 64, C_NPIX, lane);
  load_rows_f16_T(V + kvbase, &sVT[0][0], 64, C_NPIX, lane);
  __syncthreads();

  // S = (Q x K^T) * scale
#pragma unroll
  for (int mt = 0; mt < 3; ++mt)
#pragma unroll
    for (int nt = 0; nt < 4; ++nt) {
      v8f acc = {};
#pragma unroll
      for (int ks = 0; ks < 2; ++ks) {
        v16h a  = load_a_frag(&sQ[0][0], 64, mt * 16, ks * 32, lane);
        v16h bb = load_b_frag_t(&sK[0][0], 64, ks * 32, nt * 16, lane);
        acc = __builtin_amdgcn_wmma_f32_16x16x32_f16(false, a, false, bb, (short)0, acc, false, false);
      }
      const int c = nt * 16 + (lane & 15);
      const int rb = mt * 16 + ((lane & 16) ? 8 : 0);
#pragma unroll
      for (int j = 0; j < 8; ++j) sS[rb + j][c] = acc[j] * 0.125f;
    }
  __syncthreads();

  // row-gated softmax over 49 keys, then MASK2 (rows < 2 -> 0)
  for (int r = lane; r < 48; r += 32) {
    if (r < C_T) {
      const float gt = gate[(size_t)bh * C_T + r];
      float mx = -3.4e38f;
      for (int j = 0; j < C_NPIX; ++j) { float s = sS[r][j] * gt; sS[r][j] = s; mx = fmaxf(mx, s); }
      float sum = 0.f;
      for (int j = 0; j < C_NPIX; ++j) { float e = __expf(sS[r][j] - mx); sS[r][j] = e; sum += e; }
      const float norm = (r >= 2) ? (1.f / sum) : 0.f;
      for (int j = 0; j < 64; ++j)
        sP[r][j] = (j < C_NPIX) ? (_Float16)(sS[r][j] * norm) : (_Float16)0.f;
    } else {
      for (int j = 0; j < 64; ++j) sP[r][j] = (_Float16)0.f;
    }
  }
  __syncthreads();

  // O = P x V
#pragma unroll
  for (int mt = 0; mt < 3; ++mt)
#pragma unroll
    for (int nt = 0; nt < 4; ++nt) {
      v8f acc = {};
#pragma unroll
      for (int ks = 0; ks < 2; ++ks) {
        v16h a  = load_a_frag(&sP[0][0], 64, mt * 16, ks * 32, lane);
        v16h bb = load_b_frag_t(&sVT[0][0], 64, ks * 32, nt * 16, lane);
        acc = __builtin_amdgcn_wmma_f32_16x16x32_f16(false, a, false, bb, (short)0, acc, false, false);
      }
      const int c = nt * 16 + (lane & 15);
      const int rb = mt * 16 + ((lane & 16) ? 8 : 0);
#pragma unroll
      for (int j = 0; j < 8; ++j) {
        const int r = rb + j;
        if (r < C_T) O[qbase + (size_t)r * 512 + c] = acc[j];
      }
    }
}

// ---------------- host-side launch ----------------
static inline void launch_gemm(const float* A, const float* W, const float* bias,
                               float* C, int M, int N, int K, int relu,
                               hipStream_t stream) {
  dim3 grid((N + BN - 1) / BN, (M + BM - 1) / BM);
  gemm_wmma_kernel<<<grid, 256, 0, stream>>>(A, W, bias, C, M, N, K, relu);
}

extern "C" void kernel_launch(void* const* d_in, const int* in_sizes, int n_in,
                              void* d_out, int out_size, void* d_ws, size_t ws_size,
                              hipStream_t stream) {
  const float* headlines = (const float*)d_in[0];
  const float* abstracts = (const float*)d_in[1];
  const float* images    = (const float*)d_in[2];
  const float* captions  = (const float*)d_in[3];
  const float* bodies    = (const float*)d_in[4];
  const float* Wh = (const float*)d_in[5];  const float* bh = (const float*)d_in[6];
  const float* Wa = (const float*)d_in[7];  const float* ba = (const float*)d_in[8];
  const float* Wi = (const float*)d_in[9];  const float* bi = (const float*)d_in[10];
  const float* Wc = (const float*)d_in[11]; const float* bc = (const float*)d_in[12];
  const float* Wb = (const float*)d_in[13]; const float* bb = (const float*)d_in[14];
  const float* Wq_t = (const float*)d_in[15]; const float* bq_t = (const float*)d_in[16];
  const float* Wk_t = (const float*)d_in[17]; const float* bk_t = (const float*)d_in[18];
  const float* Wv_t = (const float*)d_in[19]; const float* bv_t = (const float*)d_in[20];
  const float* Wfc_t = (const float*)d_in[21]; const float* bfc_t = (const float*)d_in[22];
  const float* g_t = (const float*)d_in[23];  const float* be_t = (const float*)d_in[24];
  const float* Wq_g = (const float*)d_in[25]; const float* bq_g = (const float*)d_in[26];
  const float* Wk_g = (const float*)d_in[27]; const float* bk_g = (const float*)d_in[28];
  const float* Wv_g = (const float*)d_in[29]; const float* bv_g = (const float*)d_in[30];
  const float* Wfc_g = (const float*)d_in[31]; const float* bfc_g = (const float*)d_in[32];
  const float* g_g = (const float*)d_in[33];  const float* be_g = (const float*)d_in[34];
  const float* W1c = (const float*)d_in[35]; const float* b1c = (const float*)d_in[36];
  const float* W2c = (const float*)d_in[37]; const float* b2c = (const float*)d_in[38];
  const float* W1a = (const float*)d_in[39]; const float* b1a = (const float*)d_in[40];
  const float* W2a = (const float*)d_in[41]; const float* b2a = (const float*)d_in[42];

  const int B  = in_sizes[0] / 768;
  const int MT = B * C_T;          // time rows
  const int MP = B * C_NPIX;       // pixel rows

  float* ws = (float*)d_ws;
  size_t off = 0;
  auto alloc = [&](size_t n) { float* p = ws + off; off += n; return p; };

  float* tmp_h   = alloc((size_t)B * C_DM);
  float* tmp_a   = alloc((size_t)B * C_DM);
  float* tmp_c   = alloc((size_t)B * C_DM);
  float* spatial = alloc((size_t)MP * C_DM);
  float* tmp_b   = alloc((size_t)B * 40 * C_DM);
  float* imean   = alloc((size_t)B * C_DM);
  float* tcache  = alloc((size_t)MT * C_DM);
  float* bufQ    = alloc((size_t)MT * 512);
  float* bufK    = alloc((size_t)MP * 512);
  float* bufV    = alloc((size_t)MP * 512);
  float* bufAO   = alloc((size_t)MT * 512);
  float* bufO1   = alloc((size_t)MT * C_DM);
  float* bufO2   = alloc((size_t)MT * C_DM);
  float* bufGate = alloc((size_t)B * C_H * C_T);
  float* bufFFN  = alloc((size_t)MT * C_DFC);
  (void)ws_size;

  // ---- input projections ----
  launch_gemm(headlines, Wh, bh, tmp_h,   B,      C_DM, 768, 0, stream);
  launch_gemm(abstracts, Wa, ba, tmp_a,   B,      C_DM, 768, 0, stream);
  launch_gemm(captions,  Wc, bc, tmp_c,   B,      C_DM, 768, 0, stream);
  launch_gemm(images,    Wi, bi, spatial, MP,     C_DM, 512, 0, stream);
  launch_gemm(bodies,    Wb, bb, tmp_b,   B * 40, C_DM, 768, 0, stream);

  img_mean_kernel<<<B, 128, 0, stream>>>(spatial, imean);
  assemble_tc_kernel<<<MT, 128, 0, stream>>>(tmp_h, tmp_a, imean, tmp_c, tmp_b, tcache);

  // ---- self-attention block ----
  launch_gemm(tcache, Wq_t, bq_t, bufQ, MT, 512, C_DM, 0, stream);
  launch_gemm(tcache, Wk_t, bk_t, bufK, MT, 512, C_DM, 0, stream);
  launch_gemm(tcache, Wv_t, bv_t, bufV, MT, 512, C_DM, 0, stream);
  attn_self_kernel<<<B * C_H, 32, 0, stream>>>(bufQ, bufK, bufV, bufAO, bufGate);
  launch_gemm(bufAO, Wfc_t, bfc_t, bufO1, MT, C_DM, 512, 0, stream);
  add_ln_kernel<<<MT, 128, 0, stream>>>(bufO1, tcache, g_t, be_t, bufO1);

  // ---- gated cross-attention block ----
  launch_gemm(bufO1,   Wq_g, bq_g, bufQ, MT, 512, C_DM, 0, stream);
  launch_gemm(spatial, Wk_g, bk_g, bufK, MP, 512, C_DM, 0, stream);
  launch_gemm(spatial, Wv_g, bv_g, bufV, MP, 512, C_DM, 0, stream);
  attn_cross_kernel<<<B * C_H, 32, 0, stream>>>(bufQ, bufK, bufV, bufGate, bufAO);
  launch_gemm(bufAO, Wfc_g, bfc_g, bufO2, MT, C_DM, 512, 0, stream);
  add_ln_kernel<<<MT, 128, 0, stream>>>(bufO2, bufO1, g_g, be_g, bufO2);

  // ---- classification / action heads ----
  float* out_class = (float*)d_out;
  float* out_act   = out_class + (size_t)MT * C_NC;
  launch_gemm(bufO2,  W1c, b1c, bufFFN,    MT, C_DFC, C_DM,  1, stream);
  launch_gemm(bufFFN, W2c, b2c, out_class, MT, C_NC,  C_DFC, 0, stream);
  launch_gemm(bufO2,  W1a, b1a, bufFFN,    MT, C_DFC, C_DM,  1, stream);
  launch_gemm(bufFFN, W2a, b2a, out_act,   MT, C_NA,  C_DFC, 0, stream);
}
